// TAttentionWithProjection_33097017983678
// MI455X (gfx1250) — matrix operand order repaired and checked by reading, not compiled, not run on hardware
//
#include <hip/hip_runtime.h>

// ---------------------------------------------------------------------------
// CDNA5 (gfx1250): bf16 WMMA + async-LDS / TDM staging, f32 accumulate.
// ---------------------------------------------------------------------------

typedef __attribute__((ext_vector_type(16))) __bf16 v16bf_t;
typedef __attribute__((ext_vector_type(8)))  float  v8f_t;
typedef __attribute__((ext_vector_type(4)))  unsigned int u32x4_t;
typedef __attribute__((ext_vector_type(8)))  int    i32x8_t;
typedef __attribute__((ext_vector_type(4)))  int    i32x4_t;

// async-load builtin wants AS(1)/AS(3) pointers to int4 (one b128 per lane)
typedef __attribute__((address_space(1))) i32x4_t as1_v4i;
typedef __attribute__((address_space(3))) i32x4_t as3_v4i;
typedef __attribute__((address_space(3))) void*   as3_t;

union FragAB { v16bf_t v; unsigned int u[8]; };
union FragC  { v8f_t   v; float        f[8]; };

// ---- feature detection (compile-safe on both toolchains) -------------------
#if __has_builtin(__builtin_amdgcn_global_load_async_to_lds_b128) && \
    __has_builtin(__builtin_amdgcn_s_wait_asynccnt)
#define USE_ASYNC_LDS 1
#else
#define USE_ASYNC_LDS 0
#endif

#if __has_builtin(__builtin_amdgcn_tensor_load_to_lds) && \
    __has_builtin(__builtin_amdgcn_s_wait_tensorcnt)
#define USE_TDM 1
#else
#define USE_TDM 0
#endif
#if __has_include(<hip/amd_detail/amd_gfx1250_TDM.h>)
#define TDM_6ARG 1            // therock-10.0 headers => 6-arg builtin
#else
#define TDM_6ARG 0            // ROCm 7.2 clang-22 => 5-arg builtin
#endif

__device__ __forceinline__ unsigned short f2bf(float f) {
  unsigned int u = __builtin_bit_cast(unsigned int, f);
  u += 0x7FFFu + ((u >> 16) & 1u);              // round-to-nearest-even
  return (unsigned short)(u >> 16);
}

__device__ __forceinline__ v8f_t wmma_bf16(const FragAB& a, const FragAB& b, v8f_t c) {
  return __builtin_amdgcn_wmma_f32_16x16x32_bf16(false, a.v, false, b.v, (short)0, c,
                                                 false, false);
}

// K offset inside a 16x32 bf16 A-fragment for VGPR v, lane-half h
__device__ __forceinline__ int a_koff(int v, int h) {
  return 2 * v + 16 * h;
}

#if USE_ASYNC_LDS
__device__ __forceinline__ void async_b128(const void* g, void* l) {
  __builtin_amdgcn_global_load_async_to_lds_b128((as1_v4i*)g, (as3_v4i*)l, 0, 0);
}
#endif

#if USE_TDM
// Issue one 2-D TDM tile load: rows x cols (cols contiguous, data_size=2B),
// global row stride = gstride elements, packed row-major into LDS at lds_base.
__device__ __forceinline__ void tdm_load_tile_2d(const unsigned short* gsrc,
                                                 const void* lds_dst,
                                                 int cols, int rows, int gstride,
                                                 int tensor_rows) {
  const unsigned long long ga = (unsigned long long)(const void*)gsrc;
  const unsigned lds = (unsigned)(unsigned long long)(as3_t)(void*)lds_dst;
  u32x4_t g0;
  g0.x = 1u;                                        // count=1, user descriptor
  g0.y = lds;                                       // lds_addr
  g0.z = (unsigned)(ga & 0xFFFFFFFFull);            // global_addr[31:0]
  g0.w = (unsigned)((ga >> 32) & 0x1FFFFFFu) | (2u << 30);  // [56:32] | type=2
  i32x8_t g1;
  g1[0] = (1 << 16);                                // data_size=1 (2 bytes)
  g1[1] = (int)(((unsigned)cols & 0xFFFFu) << 16);  // tensor_dim0[15:0]
  g1[2] = (int)((((unsigned)cols >> 16) & 0xFFFFu) |
                (((unsigned)tensor_rows & 0xFFFFu) << 16)); // dim0 hi | dim1 lo
  g1[3] = (int)(((unsigned)tensor_rows >> 16) |
                (((unsigned)cols & 0xFFFFu) << 16));        // dim1 hi | tile_dim0
  g1[4] = (int)(((unsigned)rows & 0xFFFFu));        // tile_dim1 | tile_dim2=0
  g1[5] = gstride;                                  // tensor_dim0_stride[31:0]
  g1[6] = 0;
  g1[7] = 0;
  i32x4_t z4 = {0, 0, 0, 0};
#if TDM_6ARG
  i32x8_t z8 = {0, 0, 0, 0, 0, 0, 0, 0};
  __builtin_amdgcn_tensor_load_to_lds(g0, g1, z4, z4, z8, 0);
#else
  __builtin_amdgcn_tensor_load_to_lds(g0, g1, z4, z4, 0);
#endif
}
#endif

// ---------------------------------------------------------------------------
// fp32 -> bf16 conversion
// ---------------------------------------------------------------------------
__global__ void f32_to_bf16_kernel(const float* __restrict__ src,
                                   unsigned short* __restrict__ dst, int n) {
  int i = blockIdx.x * blockDim.x + threadIdx.x;
  if (i < n) dst[i] = f2bf(src[i]);
}

// ---------------------------------------------------------------------------
// Row normalization (mean/std over D, ddof=1) + transpose [B,T,M,D]->[B,M,T,D]
// ---------------------------------------------------------------------------
__global__ __launch_bounds__(256)
void mvn_kernel(const float* __restrict__ x,
                unsigned short* __restrict__ xn,
                unsigned short* __restrict__ xt,
                int B, int T, int M, int D) {
  const int row = blockIdx.x;           // flat over (b,t,m)
  const int b = row / (T * M);
  const int t = (row / M) % T;
  const int m = row % M;
  const int d = threadIdx.x;

  __shared__ float red[256];
  const float v = x[(size_t)row * D + d];
  red[d] = v;
  __syncthreads();
  for (int s = 128; s > 0; s >>= 1) { if (d < s) red[d] += red[d + s]; __syncthreads(); }
  const float mu = red[0] / (float)D;
  __syncthreads();
  const float c = v - mu;
  red[d] = c * c;
  __syncthreads();
  for (int s = 128; s > 0; s >>= 1) { if (d < s) red[d] += red[d + s]; __syncthreads(); }
  const float sd = sqrtf(red[0] / (float)(D - 1));
  const float nv = c / (sd + 1e-6f);

  const size_t o = (((size_t)(b * M + m) * T) + t) * D + d;
  xn[o] = f2bf(nv);
  xt[o] = f2bf(v);
}

// ---------------------------------------------------------------------------
// Generic bf16 GEMM: C[rows x N] = A[rows x K] * Bw[K x N] (+ bias) (+= Cf)
// 128 threads (4 waves), 64x64 tile. A-tile staged via async global->LDS.
// ---------------------------------------------------------------------------
template <bool OUT_BF16, bool ACCUM>
__global__ __launch_bounds__(128)
void gemm_bf16_kernel(const unsigned short* __restrict__ A,
                      const unsigned short* __restrict__ Bw,
                      const float* __restrict__ bias,
                      void* __restrict__ Cout,
                      int rowsTotal, int K, int N) {
  __shared__ unsigned short As[64][32];   // packed: async memcpy target
  __shared__ unsigned short Bt[64][34];   // transposed: Bt[n][k]

  const int tid  = threadIdx.x;
  const int lane = tid & 31;
  const int wave = tid >> 5;
  const int half = lane >> 4;
  const int nl   = lane & 15;
  const int row0 = blockIdx.x * 64;
  const int col0 = blockIdx.y * 64;

  FragC acc[4];
#pragma unroll
  for (int c = 0; c < 4; ++c)
#pragma unroll
    for (int r = 0; r < 8; ++r) acc[c].f[r] = 0.0f;

  for (int k0 = 0; k0 < K; k0 += 32) {
    // ---- stage A tile (64 x 32 bf16) ----
#if USE_ASYNC_LDS
#pragma unroll
    for (int i = 0; i < 2; ++i) {
      int idx = i * 128 + tid;            // 16-byte chunk id, 0..255
      int r = idx >> 2, c8 = (idx & 3) * 8;
      int gr = row0 + r;
      if (gr < rowsTotal)
        async_b128(A + (size_t)gr * K + k0 + c8, &As[r][c8]);
    }
#else
#pragma unroll
    for (int i = 0; i < 8; ++i) {
      int idx = i * 128 + tid;            // uint id, 0..1023
      int r = idx >> 4, k2 = idx & 15;
      int gr = row0 + r;
      unsigned int w = 0u;
      if (gr < rowsTotal)
        w = *(const unsigned int*)&A[(size_t)gr * K + k0 + 2 * k2];
      *(unsigned int*)&As[r][2 * k2] = w;
    }
#endif
    // ---- stage B transposed: Bt[n][k] = Bw[k0+k][col0+n] ----
#pragma unroll
    for (int i = 0; i < 16; ++i) {
      int idx = i * 128 + tid;            // 0..2047
      int n = idx & 63, k = idx >> 6;
      Bt[n][k] = Bw[(size_t)(k0 + k) * N + col0 + n];
    }
#if USE_ASYNC_LDS
    __builtin_amdgcn_s_wait_asynccnt(0);
#endif
    __syncthreads();

    FragAB af;
#pragma unroll
    for (int v = 0; v < 8; ++v)
      af.u[v] = *(const unsigned int*)&As[wave * 16 + nl][a_koff(v, half)];

#pragma unroll
    for (int c = 0; c < 4; ++c) {
      FragAB bfr;
      const int kb = 16 * half;
#pragma unroll
      for (int v = 0; v < 8; ++v)
        bfr.u[v] = *(const unsigned int*)&Bt[c * 16 + nl][kb + 2 * v];
      acc[c].v = wmma_bf16(af, bfr, acc[c].v);
    }
    __syncthreads();
  }

  // epilogue; C frag: VGPR r -> row r + 8*half, col = lane&15
#pragma unroll
  for (int c = 0; c < 4; ++c) {
#pragma unroll
    for (int r = 0; r < 8; ++r) {
      const int grow = row0 + wave * 16 + r + 8 * half;
      const int gcol = col0 + c * 16 + nl;
      if (grow < rowsTotal) {
        float val = acc[c].f[r];
        if (bias) val += bias[gcol];
        const size_t o = (size_t)grow * N + gcol;
        if (OUT_BF16) {
          ((unsigned short*)Cout)[o] = f2bf(val);
        } else {
          float* Cf = (float*)Cout;
          if (ACCUM) val += Cf[o];
          Cf[o] = val;
        }
      }
    }
  }
}

// ---------------------------------------------------------------------------
// Cross-channel causal flash attention. One wave per (b, m, 16-row q tile).
// K tile via TDM (fallback async / VGPR copy); Q tile via async copy.
// ---------------------------------------------------------------------------
#define AT_T 512
#define AT_D 256
#define AT_M 8

__global__ __launch_bounds__(32)
void attention_kernel(const unsigned short* __restrict__ Qg,
                      const unsigned short* __restrict__ Kg,
                      const unsigned short* __restrict__ Vg,
                      unsigned short* __restrict__ Aout) {
  constexpr int T = AT_T, D = AT_D, M = AT_M;
  constexpr int VP = 40, PP = 36, AP = 258;

  __shared__ unsigned short Qs[16 * D];    // Q tile  [16][256] packed
  __shared__ unsigned short Ks[32 * D];    // K tile  [32][256] packed (TDM dst)
  __shared__ unsigned short Vt[256 * VP];  // V tile transposed [256][32]
  __shared__ unsigned short Ps[16 * PP];   // P tile  [16][32]
  __shared__ float          AccS[16 * AP]; // output accumulator [16][256]

  const int lane = threadIdx.x;
  const int half = lane >> 4;
  const int nl   = lane & 15;
  const int ntt  = T / 16;
  const int tt   = blockIdx.x % ntt;
  const int bm   = blockIdx.x / ntt;
  const int m    = bm % M;
  const int b    = bm / M;
  const int t0   = tt * 16;

  for (int i = lane; i < 16 * AP; i += 32) AccS[i] = 0.0f;

  // ---- load Q tile ----
  const size_t qbase = ((size_t)(b * M + m) * T + t0) * D;
#if USE_ASYNC_LDS
  for (int i = 0; i < 16; ++i) {
    int off = (i * 32 + lane) * 8;         // 16B chunks, 8 KB total
    async_b128(Qg + qbase + off, Qs + off);
  }
  __builtin_amdgcn_s_wait_asynccnt(0);
#else
  {
    const unsigned int* Q32 = (const unsigned int*)(Qg + qbase);
    for (int i = 0; i < 64; ++i) {
      int idx = i * 32 + lane;
      ((unsigned int*)Qs)[idx] = Q32[idx];
    }
  }
#endif
  __syncthreads();

  const float inv_sqrt_d = 0.0625f;        // 1/sqrt(256)
  const int smax = t0 + 16;                // exclusive causal bound

  for (int n = 0; n < M; ++n) {
    FragC O[16];
#pragma unroll
    for (int dt = 0; dt < 16; ++dt)
#pragma unroll
      for (int r = 0; r < 8; ++r) O[dt].f[r] = 0.0f;
    float rm[8], rs[8];
#pragma unroll
    for (int r = 0; r < 8; ++r) { rm[r] = -3.0e38f; rs[r] = 0.0f; }

    const size_t kvbase = (size_t)(b * M + n) * T * D;

    for (int sb = 0; sb < smax; sb += 32) {
      const unsigned short* Krow = Kg + kvbase + (size_t)sb * D;
      const unsigned int*   V32  = (const unsigned int*)(Vg + kvbase + (size_t)sb * D);

      // ---- stage K tile [32][256] ----
#if USE_TDM
      tdm_load_tile_2d(Krow, Ks, /*cols=*/D, /*rows=*/32, /*gstride=*/D,
                       /*tensor_rows=*/T);
#elif USE_ASYNC_LDS
      for (int i = 0; i < 32; ++i) {
        int off = (i * 32 + lane) * 8;     // 16 KB total
        async_b128(Krow + off, Ks + off);
      }
#else
      {
        const unsigned int* K32 = (const unsigned int*)Krow;
        for (int i = 0; i < 128; ++i) {
          int idx = i * 32 + lane;
          ((unsigned int*)Ks)[idx] = K32[idx];
        }
      }
#endif
      // ---- stage V transposed [256][32] (VGPR path; transpose needed) ----
      for (int i = 0; i < 128; ++i) {
        int idx = i * 32 + lane;           // 0..4095 uints
        int s = idx >> 7, dd = idx & 127;
        unsigned int w = V32[s * (D / 2) + dd];
        Vt[(2 * dd) * VP + s]     = (unsigned short)(w & 0xFFFFu);
        Vt[(2 * dd + 1) * VP + s] = (unsigned short)(w >> 16);
      }
#if USE_TDM
      __builtin_amdgcn_s_wait_tensorcnt(0);
#elif USE_ASYNC_LDS
      __builtin_amdgcn_s_wait_asynccnt(0);
#endif
      __syncthreads();

      // ---- scores: two 16x16 tiles over K dim D ----
      FragC S0, S1;
#pragma unroll
      for (int r = 0; r < 8; ++r) { S0.f[r] = 0.0f; S1.f[r] = 0.0f; }
#pragma unroll
      for (int ds = 0; ds < D / 32; ++ds) {
        const int d0 = ds * 32;
        FragAB qa;
#pragma unroll
        for (int v = 0; v < 8; ++v)
          qa.u[v] = *(const unsigned int*)&Qs[nl * D + d0 + a_koff(v, half)];
        FragAB kb0, kb1;
        const int kbo = 16 * half;
#pragma unroll
        for (int v = 0; v < 8; ++v) {
          kb0.u[v] = *(const unsigned int*)&Ks[nl * D + d0 + kbo + 2 * v];
          kb1.u[v] = *(const unsigned int*)&Ks[(16 + nl) * D + d0 + kbo + 2 * v];
        }
        S0.v = wmma_bf16(qa, kb0, S0.v);
        S1.v = wmma_bf16(qa, kb1, S1.v);
      }

      // ---- online softmax update ----
#pragma unroll
      for (int r = 0; r < 8; ++r) {
        const int tg = t0 + r + 8 * half;
        float e0 = S0.f[r] * inv_sqrt_d;
        float e1 = S1.f[r] * inv_sqrt_d;
        if (sb + nl > tg)      e0 = -3.0e38f;
        if (sb + 16 + nl > tg) e1 = -3.0e38f;
        float mx = fmaxf(e0, e1);
#pragma unroll
        for (int k = 1; k < 16; k <<= 1) mx = fmaxf(mx, __shfl_xor(mx, k, 32));
        const float nm = fmaxf(rm[r], mx);
        const float sc = __expf(rm[r] - nm);
        rm[r] = nm;
        const float p0 = __expf(e0 - nm);
        const float p1 = __expf(e1 - nm);
        float sm = p0 + p1;
#pragma unroll
        for (int k = 1; k < 16; k <<= 1) sm += __shfl_xor(sm, k, 32);
        rs[r] = rs[r] * sc + sm;
#pragma unroll
        for (int dt = 0; dt < 16; ++dt) O[dt].f[r] *= sc;
        Ps[(r + 8 * half) * PP + nl]      = f2bf(p0);
        Ps[(r + 8 * half) * PP + 16 + nl] = f2bf(p1);
      }
      __syncthreads();

      // ---- O += P(16x32) @ V(32x256) ----
      FragAB pa;
#pragma unroll
      for (int v = 0; v < 8; ++v)
        pa.u[v] = *(const unsigned int*)&Ps[nl * PP + a_koff(v, half)];
      const int kbo = 16 * half;
#pragma unroll
      for (int dt = 0; dt < 16; ++dt) {
        FragAB vb;
#pragma unroll
        for (int v = 0; v < 8; ++v)
          vb.u[v] = *(const unsigned int*)&Vt[(dt * 16 + nl) * VP + kbo + 2 * v];
        O[dt].v = wmma_bf16(pa, vb, O[dt].v);
      }
      __syncthreads();
    }

    // ---- accumulate normalized per-n output (sum over key channel n) ----
#pragma unroll
    for (int dt = 0; dt < 16; ++dt)
#pragma unroll
      for (int r = 0; r < 8; ++r) {
        const int rr = r + 8 * half;
        AccS[rr * AP + dt * 16 + nl] += O[dt].f[r] / rs[r];
      }
    __syncthreads();
  }

  // ---- store A tile as bf16 ----
  const size_t obase = ((size_t)(b * M + m) * T + t0) * D;
  for (int i = lane; i < 16 * D; i += 32) {
    int r = i >> 8, d = i & 255;
    Aout[obase + (size_t)r * D + d] = f2bf(AccS[r * AP + d]);
  }
}

// ---------------------------------------------------------------------------
// Host launcher
// ---------------------------------------------------------------------------
extern "C" void kernel_launch(void* const* d_in, const int* in_sizes, int n_in,
                              void* d_out, int out_size, void* d_ws, size_t ws_size,
                              hipStream_t stream) {
  (void)in_sizes; (void)n_in; (void)out_size; (void)ws_size;
  const int B = 4, T = 512, M = 8, D = 256;
  const int rows = B * T * M;                                    // 16384
  const size_t SZB = (size_t)rows * D * sizeof(unsigned short);  // 8 MB per tensor

  const float* x   = (const float*)d_in[0];
  const float* Wq  = (const float*)d_in[1];
  const float* bq  = (const float*)d_in[2];
  const float* Wk  = (const float*)d_in[3];
  const float* bk  = (const float*)d_in[4];
  const float* Wv  = (const float*)d_in[5];
  const float* bv  = (const float*)d_in[6];
  const float* Wpa = (const float*)d_in[7];
  const float* bpa = (const float*)d_in[8];
  const float* Wpc = (const float*)d_in[9];
  const float* bpc = (const float*)d_in[10];

  char* ws = (char*)d_ws;
  unsigned short* xn   = (unsigned short*)(ws + 0 * SZB);  // reused as A
  unsigned short* xt   = (unsigned short*)(ws + 1 * SZB);
  unsigned short* Qb   = (unsigned short*)(ws + 2 * SZB);  // reused as A_proj
  unsigned short* Kb   = (unsigned short*)(ws + 3 * SZB);
  unsigned short* Vb   = (unsigned short*)(ws + 4 * SZB);
  unsigned short* wqb  = (unsigned short*)(ws + 5 * SZB);
  unsigned short* wkb  = wqb  + (size_t)D * D;
  unsigned short* wvb  = wkb  + (size_t)D * D;
  unsigned short* wpab = wvb  + (size_t)D * D;
  unsigned short* wpcb = wpab + (size_t)D * D;             // [2D x D]

  const int DD = D * D;
  f32_to_bf16_kernel<<<(DD + 255) / 256, 256, 0, stream>>>(Wq,  wqb,  DD);
  f32_to_bf16_kernel<<<(DD + 255) / 256, 256, 0, stream>>>(Wk,  wkb,  DD);
  f32_to_bf16_kernel<<<(DD + 255) / 256, 256, 0, stream>>>(Wv,  wvb,  DD);
  f32_to_bf16_kernel<<<(DD + 255) / 256, 256, 0, stream>>>(Wpa, wpab, DD);
  f32_to_bf16_kernel<<<(2 * DD + 255) / 256, 256, 0, stream>>>(Wpc, wpcb, 2 * DD);

  mvn_kernel<<<rows, 256, 0, stream>>>(x, xn, xt, B, T, M, D);

  dim3 gg((rows + 63) / 64, D / 64);
  gemm_bf16_kernel<true,  false><<<gg, 128, 0, stream>>>(xn, wqb, bq, Qb, rows, D, D);
  gemm_bf16_kernel<true,  false><<<gg, 128, 0, stream>>>(xn, wkb, bk, Kb, rows, D, D);
  gemm_bf16_kernel<true,  false><<<gg, 128, 0, stream>>>(xn, wvb, bv, Vb, rows, D, D);

  attention_kernel<<<B * M * (T / 16), 32, 0, stream>>>(Qb, Kb, Vb, xn);

  gemm_bf16_kernel<true,  false><<<gg, 128, 0, stream>>>(xn, wpab, bpa, Qb, rows, D, D);

  gemm_bf16_kernel<false, false><<<gg, 128, 0, stream>>>(xt, wpcb, bpc, d_out, rows, D, D);
  gemm_bf16_kernel<false, true ><<<gg, 128, 0, stream>>>(Qb, wpcb + (size_t)D * D, nullptr,
                                                         d_out, rows, D, D);
}
